// FlashAttention_87007447482440
// MI455X (gfx1250) — matrix operand order repaired and checked by reading, not compiled, not run on hardware
//
#include <hip/hip_runtime.h>
#include <hip/hip_bf16.h>
#include <math.h>

typedef __attribute__((ext_vector_type(16))) _Float16 v16h;
typedef __attribute__((ext_vector_type(8)))  _Float16 v8h;
typedef __attribute__((ext_vector_type(8)))  float    v8f;
typedef __attribute__((ext_vector_type(4)))  unsigned v4u;
typedef __attribute__((ext_vector_type(8)))  int      v8i;
typedef __attribute__((ext_vector_type(4)))  int      v4i;

#define D_MODEL 1024
#define NHEADS  16
#define DK      64
#define BATCH   2
#define SEQ     2048
#define MTOT    (BATCH * SEQ)   // 4096 rows for all projection GEMMs

// ---------------------------------------------------------------------------
// Fragment helpers (wave32 WMMA 16x16x32 f16 layouts, CDNA5 ISA)
// A 16x32 f16: lane m=l&15; lanes 0-15 hold K {0..7,16..23}, lanes 16-31 hold
// K {8..15,24..31}; two contiguous 8-half runs -> two b128 loads.
// B 32x16 f16: lane n=l&15; lanes 0-15 hold K {0..7,16..23}, lanes 16-31 hold
// K {8..15,24..31} (same as A) -> two b128 loads.
// C/D 16x16 f32: VGPR r, lanes 0-15 -> (M=r, N=lane); lanes 16-31 -> (M=r+8).
// ---------------------------------------------------------------------------
__device__ __forceinline__ v16h cat16(v8h a, v8h b) {
  return __builtin_shufflevector(a, b, 0,1,2,3,4,5,6,7,8,9,10,11,12,13,14,15);
}

__device__ __forceinline__ v16h load_a_frag(const _Float16* rowptr, int lane) {
  const int k0 = (lane < 16) ? 0 : 8;
  v8h lo = *reinterpret_cast<const v8h*>(rowptr + k0);
  v8h hi = *reinterpret_cast<const v8h*>(rowptr + k0 + 16);
  return cat16(lo, hi);
}

__device__ __forceinline__ v16h load_b_frag(const _Float16* colptr, int lane) {
  const int k0 = (lane < 16) ? 0 : 8;
  v8h lo = *reinterpret_cast<const v8h*>(colptr + k0);
  v8h hi = *reinterpret_cast<const v8h*>(colptr + k0 + 16);
  return cat16(lo, hi);
}

// ---------------------------------------------------------------------------
// Fused DPP 16-lane max reduction: v_max_num_f32 with the DPP modifier on
// src0 (1 VALU per butterfly step, no canonicalize movs). The four involutive
// permutations have xor masks {1,2,7,15}, which span all 16 lanes.
// ---------------------------------------------------------------------------
__device__ __forceinline__ float rowmax16(float x) {
  float t;
  asm("v_max_num_f32_dpp %0, %1, %1 quad_perm:[1,0,3,2] row_mask:0xf bank_mask:0xf bound_ctrl:1"
      : "=v"(t) : "v"(x));
  asm("v_max_num_f32_dpp %0, %1, %1 quad_perm:[2,3,0,1] row_mask:0xf bank_mask:0xf bound_ctrl:1"
      : "=v"(x) : "v"(t));
  asm("v_max_num_f32_dpp %0, %1, %1 row_half_mirror row_mask:0xf bank_mask:0xf bound_ctrl:1"
      : "=v"(t) : "v"(x));
  asm("v_max_num_f32_dpp %0, %1, %1 row_mirror row_mask:0xf bank_mask:0xf bound_ctrl:1"
      : "=v"(x) : "v"(t));
  return x;
}

// ---------------------------------------------------------------------------
// Tensor Data Mover: 2D f16 tile load, global -> LDS, one instruction.
// D# packing per CDNA5 ISA:
//   group0: [1:0]=count=1, [63:32]=lds_addr, [120:64]=global_addr, type=2
//   group1: data_size=1 (2B), pad fields, tensor_dim0/1, tile_dim0/1,
//           tensor_dim0_stride (48b). 2D tile -> remaining groups NULL.
// pad_interval code k -> 2^(k+1) dwords between pads; pad_amount code = dw-1.
// This toolchain exposes the 6-arg builtin (g0, g1, g2, g3, extra, cpol).
// ---------------------------------------------------------------------------
__device__ __forceinline__ void tdm_load_2d_f16(
    unsigned lds_addr, const void* gptr,
    unsigned tile_d0, unsigned tile_d1,
    unsigned tensor_d0, unsigned tensor_d1,
    unsigned d0_stride,
    unsigned pad_interval_code, unsigned pad_amount_code, bool pad_en) {
  v4u g0;
  g0[0] = 1u;                         // count=1, user mode
  g0[1] = lds_addr;                   // LDS byte address (WG-relative)
  unsigned long long ga = (unsigned long long)gptr;
  g0[2] = (unsigned)ga;                                   // addr[31:0]
  g0[3] = (unsigned)((ga >> 32) & 0x01FFFFFFull)          // addr[56:32]
          | 0x80000000u;                                  // type=2 ("image")

  v8i g1;
  g1[0] = (int)((1u << 16)                                // data_size=2B
                | (pad_en ? (1u << 20) : 0u)
                | (pad_interval_code << 22)
                | (pad_amount_code << 25));
  g1[1] = (int)((tensor_d0 & 0xFFFFu) << 16);             // tensor_dim0 lo16
  g1[2] = (int)(((tensor_d0 >> 16) & 0xFFFFu)
                | ((tensor_d1 & 0xFFFFu) << 16));         // d0 hi / d1 lo
  g1[3] = (int)(((tensor_d1 >> 16) & 0xFFFFu)
                | (tile_d0 << 16));                       // d1 hi / tile_dim0
  g1[4] = (int)(tile_d1 & 0xFFFFu);                       // tile_dim1 (2D)
  g1[5] = (int)d0_stride;                                 // dim0 stride lo32
  g1[6] = 0;                                              // stride hi, d1strd
  g1[7] = 0;

  v4i z4 = {};
  v8i z8 = {};
  __builtin_amdgcn_tensor_load_to_lds(g0, g1, z4, z4, z8, 0);
}

// ---------------------------------------------------------------------------
// Zero-fill for the 512 MB attn_weights output (memory-bound tail of d_out)
// ---------------------------------------------------------------------------
__global__ void zero_f4_kernel(float4* __restrict__ p, size_t n4) {
  size_t i = (size_t)blockIdx.x * blockDim.x + threadIdx.x;
  const size_t stride = (size_t)gridDim.x * blockDim.x;
  const float4 z = make_float4(0.f, 0.f, 0.f, 0.f);
  for (; i < n4; i += stride) p[i] = z;
}

// ---------------------------------------------------------------------------
// WMMA GEMM: Out[M=4096, N=1024] = A[4096,1024] @ W[1024,1024] + bias
// WG = 256 threads (8 waves). WG tile: 256(M) x 64(N). Wave tile: 32 x 64
// (2 A-frags + 4 B-frags -> 12 b128 LDS loads per 8 WMMAs).
// A_F16 path stages the A tile with a single TDM tensor_load_to_lds.
// ---------------------------------------------------------------------------
template <bool A_F16, bool OUT_F16_SCATTER>
__global__ __launch_bounds__(256) void gemm_wmma(const void*  __restrict__ Av,
                                                 const float* __restrict__ W,
                                                 const float* __restrict__ bias,
                                                 void* __restrict__ Out) {
  __shared__ __align__(16) _Float16 Alds[256][40];   // 256 x 32 (+4dw pad)
  __shared__ __align__(16) _Float16 Bt[64][40];      // 64(N) x 32(K) (+8 pad)

  const int tid  = threadIdx.x;
  const int wave = tid >> 5;
  const int lane = tid & 31;
  const int mbase = blockIdx.x * 256;
  const int nbase = blockIdx.y * 64;

  const float*    Af = (const float*)Av;
  const _Float16* Ah = (const _Float16*)Av;

  v8f acc[2][4] = {};

  for (int kk = 0; kk < D_MODEL; kk += 32) {
    // ---- stage A tile
    if (A_F16) {
      // TDM: 256 rows x 32 halves, row stride D_MODEL halves, LDS rows padded
      // 16 dwords + 4 dwords (codes 3/3) -> 40-half stride.
      if (wave == 0)
        tdm_load_2d_f16((unsigned)(uintptr_t)&Alds[0][0],
                        Ah + (size_t)mbase * D_MODEL + kk,
                        /*tile_d0=*/32, /*tile_d1=*/256,
                        /*tensor_d0=*/D_MODEL, /*tensor_d1=*/MTOT,
                        /*d0_stride=*/D_MODEL,
                        /*pad_interval=*/3, /*pad_amount=*/3, true);
    } else {
      const float* src = Af + (size_t)(mbase + tid) * D_MODEL + kk;
      const float4* s4 = reinterpret_cast<const float4*>(src);
#pragma unroll
      for (int j = 0; j < 4; ++j) {
        float4 x0 = s4[2 * j], x1 = s4[2 * j + 1];
        v8h hv;
        hv[0]=(_Float16)x0.x; hv[1]=(_Float16)x0.y; hv[2]=(_Float16)x0.z; hv[3]=(_Float16)x0.w;
        hv[4]=(_Float16)x1.x; hv[5]=(_Float16)x1.y; hv[6]=(_Float16)x1.z; hv[7]=(_Float16)x1.w;
        *reinterpret_cast<v8h*>(&Alds[tid][j * 8]) = hv;
      }
      if (kk + 32 < D_MODEL) __builtin_prefetch(src + 32, 0, 0);
    }
    // ---- stage W tile transposed: Bt[n][k], k in [kk,kk+32), n in [nbase,+64)
    {
      const int k  = tid >> 3;            // 0..31
      const int n0 = (tid & 7) * 8;       // 0..56
      const float* src = W + (size_t)(kk + k) * D_MODEL + nbase + n0;
      const float4* s4 = reinterpret_cast<const float4*>(src);
      float4 x0 = s4[0], x1 = s4[1];
      Bt[n0 + 0][k] = (_Float16)x0.x; Bt[n0 + 1][k] = (_Float16)x0.y;
      Bt[n0 + 2][k] = (_Float16)x0.z; Bt[n0 + 3][k] = (_Float16)x0.w;
      Bt[n0 + 4][k] = (_Float16)x1.x; Bt[n0 + 5][k] = (_Float16)x1.y;
      Bt[n0 + 6][k] = (_Float16)x1.z; Bt[n0 + 7][k] = (_Float16)x1.w;
    }
    if (A_F16 && wave == 0) __builtin_amdgcn_s_wait_tensorcnt(0);
    __syncthreads();

    const v16h a0 = load_a_frag(&Alds[wave * 32 +      (lane & 15)][0], lane);
    const v16h a1 = load_a_frag(&Alds[wave * 32 + 16 + (lane & 15)][0], lane);
#pragma unroll
    for (int t = 0; t < 4; ++t) {
      const v16h bfrag = load_b_frag(&Bt[t * 16 + (lane & 15)][0], lane);
      acc[0][t] = __builtin_amdgcn_wmma_f32_16x16x32_f16(
          false, a0, false, bfrag, (short)0, acc[0][t], false, false);
      acc[1][t] = __builtin_amdgcn_wmma_f32_16x16x32_f16(
          false, a1, false, bfrag, (short)0, acc[1][t], false, false);
    }
    __syncthreads();
  }

  // ---- epilogue: bias + store
#pragma unroll
  for (int g = 0; g < 2; ++g) {
#pragma unroll
    for (int t = 0; t < 4; ++t) {
      const int n_abs = nbase + t * 16 + (lane & 15);
      const float bv = bias[n_abs];
#pragma unroll
      for (int r = 0; r < 8; ++r) {
        const int m_abs = mbase + wave * 32 + g * 16 + r + ((lane >> 4) << 3);
        const float val = acc[g][t][r] + bv;
        if (OUT_F16_SCATTER) {
          const int b = m_abs >> 11, s = m_abs & (SEQ - 1);
          const int h = n_abs >> 6, d = n_abs & 63;
          ((_Float16*)Out)[((((size_t)b * NHEADS + h) * SEQ) + s) * DK + d] =
              (_Float16)val;
        } else {
          ((float*)Out)[(size_t)m_abs * D_MODEL + n_abs] = val;
        }
      }
    }
  }
}

// ---------------------------------------------------------------------------
// Flash attention: one WG per (b*h, 128-query block). 8 waves x 16 query rows.
// K tile staged by TDM (padded rows in hardware); V staged transposed by VALU;
// online softmax with fused-DPP row-max and WMMA row-sum.
// ---------------------------------------------------------------------------
__global__ __launch_bounds__(256) void flash_attn(const _Float16* __restrict__ Qh,
                                                  const _Float16* __restrict__ Kh,
                                                  const _Float16* __restrict__ Vh,
                                                  _Float16* __restrict__ AttnOut) {
  __shared__ __align__(16) _Float16 Klds[32][72];      // 32 keys x 64 (+4dw pad)
  __shared__ __align__(16) _Float16 Vt[64][40];        // 64 dims x 32 keys (+8)
  __shared__ __align__(16) _Float16 Plds[8][16][40];   // per-wave P 16x32 (+8)

  const int tid  = threadIdx.x;
  const int wave = tid >> 5;
  const int lane = tid & 31;
  const int bh   = blockIdx.y;          // 0..31 (b*16+h)
  const int b    = bh >> 4, h = bh & 15;
  const size_t head_off = (size_t)bh * SEQ * DK;
  const int qrow0 = blockIdx.x * 128 + wave * 16;

  // Q fragments for this wave's 16 rows (K dim = 64 -> two 16x32 chunks)
  const _Float16* qrow = Qh + head_off + (size_t)(qrow0 + (lane & 15)) * DK;
  const v16h qa0 = load_a_frag(qrow, lane);        // dims 0..31
  const v16h qa1 = load_a_frag(qrow + 32, lane);   // dims 32..63

  // all-ones B fragment: rowsum(P) = P @ Ones lands exactly in C-layout
  v16h ones;
#pragma unroll
  for (int i = 0; i < 16; ++i) ones[i] = (_Float16)1.0f;

  float mrow[8], lrow[8];
#pragma unroll
  for (int r = 0; r < 8; ++r) { mrow[r] = -1e30f; lrow[r] = 0.f; }
  v8f oacc[4] = {};
  const float scale = 0.125f;  // 1/sqrt(64)

  for (int kb = 0; kb < SEQ; kb += 32) {
    // ---- stage K block via TDM: 32 keys x 64 halves, rows padded
    // 32 dwords + 4 dwords (codes 4/3) -> 72-half LDS row stride.
    if (wave == 0)
      tdm_load_2d_f16((unsigned)(uintptr_t)&Klds[0][0],
                      Kh + head_off + (size_t)kb * DK,
                      /*tile_d0=*/DK, /*tile_d1=*/32,
                      /*tensor_d0=*/DK, /*tensor_d1=*/SEQ,
                      /*d0_stride=*/DK,
                      /*pad_interval=*/4, /*pad_amount=*/3, true);
    // ---- stage V block transposed: Vt[dim][key] (TDM cannot transpose)
    {
      const int key = tid >> 3;
      const int d0  = (tid & 7) * 8;
      const v8h vv = *reinterpret_cast<const v8h*>(
          Vh + head_off + (size_t)(kb + key) * DK + d0);
#pragma unroll
      for (int j = 0; j < 8; ++j) Vt[d0 + j][key] = vv[j];
    }
    if (wave == 0) __builtin_amdgcn_s_wait_tensorcnt(0);
    __syncthreads();

    // ---- scores S[16x32] = Q(16x64) @ K_blk^T : 2 key groups x 2 K-chunks
    v8f c0 = {}, c1 = {};
    {
      const _Float16* kr0 = &Klds[(lane & 15)][0];
      const _Float16* kr1 = &Klds[16 + (lane & 15)][0];
      const v16h b00 = load_b_frag(kr0, lane);
      const v16h b01 = load_b_frag(kr0 + 32, lane);
      const v16h b10 = load_b_frag(kr1, lane);
      const v16h b11 = load_b_frag(kr1 + 32, lane);
      c0 = __builtin_amdgcn_wmma_f32_16x16x32_f16(false, qa0, false, b00, (short)0, c0, false, false);
      c0 = __builtin_amdgcn_wmma_f32_16x16x32_f16(false, qa1, false, b01, (short)0, c0, false, false);
      c1 = __builtin_amdgcn_wmma_f32_16x16x32_f16(false, qa0, false, b10, (short)0, c1, false, false);
      c1 = __builtin_amdgcn_wmma_f32_16x16x32_f16(false, qa1, false, b11, (short)0, c1, false, false);
    }

    // ---- online softmax (per-row max via fused DPP butterflies)
    float alpha[8];
#pragma unroll
    for (int r = 0; r < 8; ++r) {
      const float s0 = c0[r] * scale;
      const float s1 = c1[r] * scale;
      const float vmax = rowmax16(fmaxf(s0, s1));
      const float mnew = fmaxf(mrow[r], vmax);
      const float a  = __expf(mrow[r] - mnew);
      const float p0 = __expf(s0 - mnew);
      const float p1 = __expf(s1 - mnew);
      mrow[r] = mnew;
      alpha[r] = a;
      lrow[r] *= a;
      // scatter P (C-layout) into wave-private LDS for A-layout reload
      const int m = r + ((lane >> 4) << 3);
      const int n = lane & 15;
      Plds[wave][m][n]      = (_Float16)p0;
      Plds[wave][m][n + 16] = (_Float16)p1;
    }
#pragma unroll
    for (int t = 0; t < 4; ++t)
#pragma unroll
      for (int r = 0; r < 8; ++r) oacc[t][r] *= alpha[r];

    // wave-private LDS round-trip: only need the DS counter, not a barrier
    asm volatile("s_wait_dscnt 0" ::: "memory");

    // ---- reload P in A-layout; row sums via the matrix pipe (P @ Ones)
    const v16h pa = load_a_frag(&Plds[wave][lane & 15][0], lane);
    {
      v8f zero = {};
      const v8f ls = __builtin_amdgcn_wmma_f32_16x16x32_f16(
          false, pa, false, ones, (short)0, zero, false, false);
#pragma unroll
      for (int r = 0; r < 8; ++r) lrow[r] += ls[r];
    }
    // ---- O += P(16x32) @ V_blk(32x64)
#pragma unroll
    for (int t = 0; t < 4; ++t) {
      const v16h vb = load_b_frag(&Vt[t * 16 + (lane & 15)][0], lane);
      oacc[t] = __builtin_amdgcn_wmma_f32_16x16x32_f16(
          false, pa, false, vb, (short)0, oacc[t], false, false);
    }
    __syncthreads();
  }

  // ---- normalize and store f16 to [B, S, D_MODEL] (head h at cols h*64..)
#pragma unroll
  for (int r = 0; r < 8; ++r) {
    const float inv = 1.0f / lrow[r];
    const int m = r + ((lane >> 4) << 3);
    const size_t rowoff =
        ((size_t)b * SEQ + (qrow0 + m)) * D_MODEL + (size_t)h * DK;
#pragma unroll
    for (int t = 0; t < 4; ++t)
      AttnOut[rowoff + t * 16 + (lane & 15)] = (_Float16)(oacc[t][r] * inv);
  }
}

// ---------------------------------------------------------------------------
extern "C" void kernel_launch(void* const* d_in, const int* in_sizes, int n_in,
                              void* d_out, int out_size, void* d_ws, size_t ws_size,
                              hipStream_t stream) {
  const float* q  = (const float*)d_in[0];
  const float* k  = (const float*)d_in[1];
  const float* v  = (const float*)d_in[2];
  const float* Wq = (const float*)d_in[3];
  const float* bq = (const float*)d_in[4];
  const float* Wk = (const float*)d_in[5];
  const float* bk = (const float*)d_in[6];
  const float* Wv = (const float*)d_in[7];
  const float* bv = (const float*)d_in[8];
  const float* Wo = (const float*)d_in[9];
  const float* bo = (const float*)d_in[10];
  float* out = (float*)d_out;

  const size_t head_elems = (size_t)BATCH * NHEADS * SEQ * DK;  // 4M halves
  _Float16* Qh = (_Float16*)d_ws;
  _Float16* Kh = Qh + head_elems;
  _Float16* Vh = Kh + head_elems;
  _Float16* Ah = Vh + head_elems;   // attention output, [4096][1024] f16

  // zero the attn_weights region (second output), 512 MB of zeros
  {
    const size_t out_main = (size_t)BATCH * SEQ * D_MODEL;
    const size_t aw = (size_t)BATCH * NHEADS * SEQ * SEQ;
    zero_f4_kernel<<<8192, 256, 0, stream>>>((float4*)(out + out_main), aw / 4);
  }

  const dim3 g(MTOT / 256, D_MODEL / 64);
  const dim3 blk(256);
  gemm_wmma<false, true><<<g, blk, 0, stream>>>(q, Wq, bq, Qh);
  gemm_wmma<false, true><<<g, blk, 0, stream>>>(k, Wk, bk, Kh);
  gemm_wmma<false, true><<<g, blk, 0, stream>>>(v, Wv, bv, Vh);

  flash_attn<<<dim3(SEQ / 128, BATCH * NHEADS), blk, 0, stream>>>(Qh, Kh, Vh, Ah);

  gemm_wmma<true, false><<<g, blk, 0, stream>>>(Ah, Wo, bo, out);
}
